// OneLayer_LSTM_88536455840381
// MI455X (gfx1250) — matrix operand order repaired and checked by reading, not compiled, not run on hardware
//
#include <hip/hip_runtime.h>
#include <hip/hip_bf16.h>

typedef __attribute__((ext_vector_type(16))) __bf16 v16bf;
typedef __attribute__((ext_vector_type(8)))  float  v8f;

#define LSEQ 2048
#define NB   64
#define NC   512
#define NH   512
#define KTOT 1024          // C + H fused K dimension
#define NWG  32            // workgroups; each owns 16 hidden columns
#define COLS 16            // hidden columns per workgroup
#define ROWS 64            // gate-slice rows per workgroup (4 gates * COLS)
#define NTHREADS 256       // 8 wave32
#define ASTR 1032          // padded LDS row stride (elements) for A tile
#define WSTR 1032          // padded LDS row stride (elements) for W slice
#define GSTR 66            // padded LDS row stride (floats) for gates tile

static __device__ __forceinline__ unsigned short f2bf(float f) {
    unsigned int u = __float_as_uint(f);
    u += 0x7FFFu + ((u >> 16) & 1u);        // round-to-nearest-even
    return (unsigned short)(u >> 16);
}
static __device__ __forceinline__ unsigned int pk2(float a, float b) {
    return (unsigned int)f2bf(a) | ((unsigned int)f2bf(b) << 16);
}

union FragU { uint4 q[2]; v16bf v; };

__global__ void lstm_ws_init(unsigned int* __restrict__ p, int n) {
    int i = blockIdx.x * blockDim.x + threadIdx.x;
    if (i < n) p[i] = 0u;
}

// one-shot bandwidth-bound pass: x f32 -> bf16 (hoists all conversion VALU work
// out of the 2048-step serial recurrence)
__global__ void x_to_bf16(const float* __restrict__ x,
                          unsigned short* __restrict__ xbf, int nunits) {
    int u = blockIdx.x * blockDim.x + threadIdx.x;
    if (u >= nunits) return;
    const float4* s = (const float4*)(x + (size_t)u * 8);
    float4 f0 = s[0], f1 = s[1];
    *((uint4*)(xbf + (size_t)u * 8)) =
        make_uint4(pk2(f0.x, f0.y), pk2(f0.z, f0.w),
                   pk2(f1.x, f1.y), pk2(f1.z, f1.w));
}

__global__ __launch_bounds__(NTHREADS, 1)
void lstm_persistent(const float* __restrict__ x,
                     const float* __restrict__ Wih,
                     const float* __restrict__ Whh,
                     const float* __restrict__ bias,
                     float* __restrict__ out,
                     int* __restrict__ cnt,
                     unsigned short* __restrict__ hbuf,
                     const unsigned short* __restrict__ xbf)
{
    // 129KB W slice + 129KB A tile + 16.5KB gates + 4KB cell state  (<320KB/WG)
    __shared__ __align__(16) unsigned short sW[ROWS * WSTR];
    __shared__ __align__(16) unsigned short sA[NB * ASTR];
    __shared__ float sG[NB * GSTR];
    __shared__ float sC[NB * COLS];

    const int tid   = threadIdx.x;
    const int hbase = blockIdx.x * COLS;
    const bool use_async = (xbf != nullptr);

    // ---- one-time: load fused [W_ih | W_hh] slice as bf16 into LDS ----
    for (int u = tid; u < ROWS * (KTOT / 8); u += NTHREADS) {
        const int n    = u >> 7;            // slice row (gate*16 + local col)
        const int kk   = (u & 127) * 8;     // K offset
        const int gate = n >> 4;
        const int grow = gate * NH + hbase + (n & 15);   // row in [4H, ...]
        const float* src = (kk < NC) ? (Wih + (size_t)grow * NC + kk)
                                     : (Whh + (size_t)grow * NH + (kk - NC));
        float4 f0 = ((const float4*)src)[0];
        float4 f1 = ((const float4*)src)[1];
        *((uint4*)&sW[n * WSTR + kk]) =
            make_uint4(pk2(f0.x, f0.y), pk2(f0.z, f0.w),
                       pk2(f1.x, f1.y), pk2(f1.z, f1.w));
    }

    // ---- per-thread pointwise cells (4 each) + cached biases ----
    float bi0[4], bf0[4], bg0[4], bo0[4];
    #pragma unroll
    for (int i = 0; i < 4; ++i) {
        const int cell = tid * 4 + i;              // = m*16 + j
        const int col  = hbase + (cell & 15);
        bi0[i] = bias[0 * NH + col];
        bf0[i] = bias[1 * NH + col];
        bg0[i] = bias[2 * NH + col];
        bo0[i] = bias[3 * NH + col];
        sC[cell] = 0.0f;                           // c0 = 0
    }

    // ---- wave tiling: 8 waves -> 4 Mtiles x 4 Ntiles (2 N-frags per wave) ----
    const int wave = tid >> 5, lane = tid & 31;
    const int mt  = wave & 3;
    const int ntp = (wave >> 2) << 1;              // 0 or 2
    const int lr  = lane & 15, kh = lane >> 4;
    const unsigned short* aP  = &sA[(mt * 16 + lr) * ASTR + kh * 8];
    const unsigned short* b0P = &sW[(ntp * 16 + lr) * WSTR + kh * 8];
    const unsigned short* b1P = b0P + 16 * WSTR;

    __syncthreads();

    for (int t = 0; t < LSEQ; ++t) {
        // ---- x half of A tile: no h dependency -> issue async copies BEFORE
        //      the inter-WG barrier, overlapping copy with sync latency ----
        if (use_async) {
            const unsigned short* xb = xbf + (size_t)t * NB * NC;
            for (int u = tid; u < NB * (NC / 8); u += NTHREADS) {
                const int m = u >> 6;
                const int kk = (u & 63) * 8;
                unsigned ldsa = (unsigned)(uintptr_t)&sA[m * ASTR + kk];
                const void* g = (const void*)(xb + m * NC + kk);
                asm volatile("global_load_async_to_lds_b128 %0, %1, off"
                             :: "v"(ldsa), "v"(g) : "memory");
            }
        }

        // ---- wait for all WGs to publish h(t) (t=0: init kernel zeroed it) ----
        if (t > 0) {
            if (tid == 0) {
                while (__hip_atomic_load(&cnt[t - 1], __ATOMIC_ACQUIRE,
                                         __HIP_MEMORY_SCOPE_AGENT) < NWG)
                    __builtin_amdgcn_s_sleep(2);
            }
            __syncthreads();
        }

        // ---- h half of A tile ----
        if (use_async) {
            const unsigned short* hr = hbuf + (t & 1) * NB * NH;
            for (int u = tid; u < NB * (NH / 8); u += NTHREADS) {
                const int m = u >> 6;
                const int kk = (u & 63) * 8;
                unsigned ldsa = (unsigned)(uintptr_t)&sA[m * ASTR + NC + kk];
                const void* g = (const void*)(hr + m * NH + kk);
                asm volatile("global_load_async_to_lds_b128 %0, %1, off"
                             :: "v"(ldsa), "v"(g) : "memory");
            }
            asm volatile("s_wait_asynccnt 0x0" ::: "memory");
        } else {
            // fallback: in-register f32->bf16 staging (no xbf workspace)
            const float* xt = x + (size_t)t * NB * NC;
            const unsigned short* hr = hbuf + (t & 1) * NB * NH;
            for (int u = tid; u < NB * (KTOT / 8); u += NTHREADS) {
                const int m  = u >> 7;
                const int kk = (u & 127) * 8;
                uint4 pkd;
                if (kk < NC) {
                    const float4* s = (const float4*)(xt + (size_t)m * NC + kk);
                    float4 f0 = s[0], f1 = s[1];
                    pkd = make_uint4(pk2(f0.x, f0.y), pk2(f0.z, f0.w),
                                     pk2(f1.x, f1.y), pk2(f1.z, f1.w));
                } else {
                    pkd = *((const uint4*)(hr + m * NH + (kk - NC)));
                }
                *((uint4*)&sA[m * ASTR + kk]) = pkd;
            }
        }
        __syncthreads();

        // ---- GEMM: gates[64 x 64] += A[64 x 1024] * Wslice^T via bf16 WMMA ----
        v8f acc0, acc1;
        #pragma unroll
        for (int i = 0; i < 8; ++i) { acc0[i] = 0.0f; acc1[i] = 0.0f; }

        #pragma unroll 4
        for (int k0 = 0; k0 < KTOT; k0 += 32) {
            FragU a, b0, b1;
            a.q[0]  = *(const uint4*)(aP  + k0);
            a.q[1]  = *(const uint4*)(aP  + k0 + 16);
            b0.q[0] = *(const uint4*)(b0P + k0);
            b0.q[1] = *(const uint4*)(b0P + k0 + 16);
            b1.q[0] = *(const uint4*)(b1P + k0);
            b1.q[1] = *(const uint4*)(b1P + k0 + 16);
            acc0 = __builtin_amdgcn_wmma_f32_16x16x32_bf16(
                       false, a.v, false, b0.v, (short)0, acc0, false, false);
            acc1 = __builtin_amdgcn_wmma_f32_16x16x32_bf16(
                       false, a.v, false, b1.v, (short)0, acc1, false, false);
        }

        // ---- scatter C fragments to gates LDS ----
        #pragma unroll
        for (int r = 0; r < 8; ++r) {
            const int row = mt * 16 + r + 8 * kh;          // batch index
            sG[row * GSTR + ntp * 16 + lr]       = acc0[r];
            sG[row * GSTR + (ntp + 1) * 16 + lr] = acc1[r];
        }
        __syncthreads();

        // ---- pointwise LSTM cell update, publish h(t+1) ----
        unsigned short* hw = hbuf + ((t + 1) & 1) * NB * NH;
        float* outT = out + (size_t)t * NB * NH;
        #pragma unroll
        for (int i = 0; i < 4; ++i) {
            const int cell = tid * 4 + i;
            const int m = cell >> 4, j = cell & 15;
            const int col = hbase + j;
            float gi = sG[m * GSTR + j]        + bi0[i];
            float gf = sG[m * GSTR + 16 + j]   + bf0[i];
            float gg = sG[m * GSTR + 32 + j]   + bg0[i];
            float go = sG[m * GSTR + 48 + j]   + bo0[i];
            float si = 1.0f / (1.0f + __expf(-gi));
            float sf = 1.0f / (1.0f + __expf(-gf));
            float so = 1.0f / (1.0f + __expf(-go));
            float eg = __expf(2.0f * gg);
            float tg = 1.0f - 2.0f / (eg + 1.0f);          // tanh(gg), inf-safe
            float c  = sf * sC[cell] + si * tg;
            sC[cell] = c;
            float ec = __expf(2.0f * c);
            float tc = 1.0f - 2.0f / (ec + 1.0f);          // tanh(c)
            float h  = so * tc;
            outT[(size_t)m * NH + col] = h;
            hw[m * NH + col] = f2bf(h);
            if (t == LSEQ - 1) {
                out[(size_t)LSEQ * NB * NH + (size_t)m * NH + col] = h;
                out[(size_t)LSEQ * NB * NH + (size_t)NB * NH + (size_t)m * NH + col] = c;
            }
        }
        __threadfence();
        __syncthreads();
        if (tid == 0)
            __hip_atomic_fetch_add(&cnt[t], 1, __ATOMIC_RELEASE,
                                   __HIP_MEMORY_SCOPE_AGENT);
    }
}

extern "C" void kernel_launch(void* const* d_in, const int* in_sizes, int n_in,
                              void* d_out, int out_size, void* d_ws, size_t ws_size,
                              hipStream_t stream) {
    const float* x   = (const float*)d_in[0];   // [L, B, C]
    const float* Wih = (const float*)d_in[1];   // [4H, C]
    const float* Whh = (const float*)d_in[2];   // [4H, H]
    const float* b   = (const float*)d_in[3];   // [4H]
    float* out = (float*)d_out;                 // outputs | h_T | c_T

    // workspace layout:
    //   [0, 8KB)                 step counters
    //   [8KB, 8KB+128KB)         double-buffered bf16 h
    //   [8KB+128KB, +128MB)      bf16 copy of x (optional, if ws large enough)
    int* cnt = (int*)d_ws;
    unsigned short* hbuf = (unsigned short*)((char*)d_ws + 8192);
    const size_t base = 8192 + (size_t)2 * NB * NH * 2;
    const size_t xbf_bytes = (size_t)LSEQ * NB * NC * 2;

    unsigned short* xbf = nullptr;
    if (ws_size >= base + xbf_bytes)
        xbf = (unsigned short*)((char*)d_ws + base);

    const int zero_words = (int)(base / 4);     // counters + both h buffers
    lstm_ws_init<<<(zero_words + 255) / 256, 256, 0, stream>>>((unsigned int*)d_ws,
                                                               zero_words);
    if (xbf) {
        const int nunits = LSEQ * NB * NC / 8;  // 8 floats per thread
        x_to_bf16<<<(nunits + 255) / 256, 256, 0, stream>>>(x, xbf, nunits);
    }
    lstm_persistent<<<NWG, NTHREADS, 0, stream>>>(x, Wih, Whh, b, out, cnt,
                                                  hbuf, xbf);
}
